// FusedLlamaLowBitDecoderlayer_48911087566905
// MI455X (gfx1250) — matrix operand order repaired and checked by reading, not compiled, not run on hardware
//
#include <hip/hip_runtime.h>
#include <hip/hip_bf16.h>

// ---------------------------------------------------------------------------
// Problem constants (from reference): B=4, S=1024, D=2048, H=16, HD=128, F=8192
// ---------------------------------------------------------------------------
constexpr int Bc  = 4;
constexpr int Sc  = 1024;
constexpr int Dc  = 2048;
constexpr int Hc  = 16;
constexpr int HDc = 128;
constexpr int Fc  = 8192;
constexpr int Mtot = Bc * Sc;          // 4096 rows in every GEMM
constexpr float LN_EPS = 1e-5f;
constexpr float ATT_SCALE = 0.08838834764831845f;   // 1/sqrt(128)

typedef __attribute__((ext_vector_type(16))) _Float16 v16h;
typedef __attribute__((ext_vector_type(8)))  float    v8f;

union Frag16 {           // 16 halves = 32 bytes = 8 VGPRs
    v16h h;
    uint4 q[2];
};

__device__ inline v8f wmma_f16(const v16h& a, const v16h& b, const v8f& c) {
    // D = A(16x32 f16) * B(32x16 f16) + C(16x16 f32)
    return __builtin_amdgcn_wmma_f32_16x16x32_f16(
        /*neg_a=*/false, a, /*neg_b=*/false, b,
        /*c_mod=*/(short)0, c, /*reuse_a=*/false, /*reuse_b=*/false);
}

// gfx1250 async copy: global memory -> LDS, 16B per lane, tracked by ASYNCcnt.
// %0 = LDS byte offset VGPR (addr[31:0] of the flat pointer = wave-relative
// LDS address per the ISA aperture rules), %1 = 64-bit global address VGPRs.
__device__ inline void async_copy_b128(void* lds_ptr, const void* gptr) {
    unsigned lds_off = (unsigned)(size_t)lds_ptr;
    asm volatile("global_load_async_to_lds_b128 %0, %1, off"
                 :: "v"(lds_off), "v"(gptr) : "memory");
}
__device__ inline void wait_async0() {
    asm volatile("s_wait_asynccnt 0" ::: "memory");
}

// Reductions across a 16-lane half of the wave (xor masks 1..8 never cross
// the 16-lane boundary, matching the C-matrix half-wave row grouping).
__device__ inline float halfmax(float v) {
    v = fmaxf(v, __shfl_xor(v, 1, 32));
    v = fmaxf(v, __shfl_xor(v, 2, 32));
    v = fmaxf(v, __shfl_xor(v, 4, 32));
    v = fmaxf(v, __shfl_xor(v, 8, 32));
    return v;
}
__device__ inline float halfsum(float v) {
    v += __shfl_xor(v, 1, 32);
    v += __shfl_xor(v, 2, 32);
    v += __shfl_xor(v, 4, 32);
    v += __shfl_xor(v, 8, 32);
    return v;
}

// ---------------------------------------------------------------------------
// f32 -> f16 conversion
// ---------------------------------------------------------------------------
__global__ void cvt_f32_to_f16(const float* __restrict__ in,
                               _Float16* __restrict__ out, size_t n) {
    size_t i = (size_t)blockIdx.x * blockDim.x + threadIdx.x;
    size_t stride = (size_t)gridDim.x * blockDim.x;
    for (; i < n; i += stride) out[i] = (_Float16)in[i];
}

// ---------------------------------------------------------------------------
// NT GEMM:  C[M,N] = A[M,K] * W[N,K]^T + bias[N]  (+ epilogue)
// A, W are f16 row-major-along-K.  Workgroup tile 128x128, 8 waves,
// each wave computes a 64x32 block of C as 4x2 WMMA 16x16 tiles.
// K tiles staged through LDS with double-buffered async copies (ASYNCcnt).
// ---------------------------------------------------------------------------
enum { EPI_QK = 0, EPI_VT = 1, EPI_RES_F32 = 2, EPI_GELU_F16 = 3 };

constexpr int LPITCH = 40;             // halves; 80B row pitch: 16B aligned,
                                       // conflict-free 16-row ds_load_b128

template <int EPI>
__global__ __launch_bounds__(256)
void gemm_nt(const _Float16* __restrict__ A, const _Float16* __restrict__ W,
             const float* __restrict__ bias, const float* __restrict__ resid,
             void* __restrict__ outp, int N, int K) {
    // [buffer][A=0 / B=1][128 rows * LPITCH halves]  -> 40 KB
    __shared__ _Float16 lsh[2][2][128 * LPITCH];

    const int t    = threadIdx.x;
    const int wave = t >> 5;
    const int lane = t & 31;
    const int half = lane >> 4;
    const int l16  = lane & 15;
    const int rowBase = blockIdx.y * 128;
    const int colBase = blockIdx.x * 128;
    const int wm = (wave >> 2) * 64;   // wave row offset inside WG tile
    const int wn = (wave & 3) * 32;    // wave col offset inside WG tile

    // Issue async copies of one 128x32 A tile + 128x32 B tile into buffer `buf`:
    // 512 16B chunks per matrix, 2 per thread per matrix.
    auto stage = [&](int buf, int k0) {
#pragma unroll
        for (int it = 0; it < 2; ++it) {
            int c  = t + it * 256;     // 0..511
            int r  = c >> 2;
            int kc = (c & 3) * 8;
            async_copy_b128(&lsh[buf][0][r * LPITCH + kc],
                            &A[(size_t)(rowBase + r) * K + k0 + kc]);
            async_copy_b128(&lsh[buf][1][r * LPITCH + kc],
                            &W[(size_t)(colBase + r) * K + k0 + kc]);
        }
    };

    v8f acc[4][2];
#pragma unroll
    for (int i = 0; i < 4; ++i)
#pragma unroll
        for (int j = 0; j < 2; ++j) acc[i][j] = (v8f){0,0,0,0,0,0,0,0};

    stage(0, 0);                        // prologue: fill buffer 0

    for (int k0 = 0; k0 < K; k0 += 32) {
        const int buf = (k0 >> 5) & 1;
        wait_async0();                  // this wave's copies have landed
        __syncthreads();                // => all waves' copies have landed;
                                        //    also: all readers of buf^1 done
        if (k0 + 32 < K) stage(buf ^ 1, k0 + 32);   // prefetch next K tile

        const _Float16* cA = lsh[buf][0];
        const _Float16* cB = lsh[buf][1];
        Frag16 af[4], bf[2];
#pragma unroll
        for (int mt = 0; mt < 4; ++mt) {
            int r = wm + mt * 16 + l16;
            af[mt].q[0] = *(const uint4*)&cA[r * LPITCH + 8 * half];
            af[mt].q[1] = *(const uint4*)&cA[r * LPITCH + 16 + 8 * half];
        }
#pragma unroll
        for (int nt = 0; nt < 2; ++nt) {
            int cgl = wn + nt * 16 + l16;
            bf[nt].q[0] = *(const uint4*)&cB[cgl * LPITCH + 8 * half];
            bf[nt].q[1] = *(const uint4*)&cB[cgl * LPITCH + 16 + 8 * half];
        }
#pragma unroll
        for (int mt = 0; mt < 4; ++mt)
#pragma unroll
            for (int nt = 0; nt < 2; ++nt)
                acc[mt][nt] = wmma_f16(af[mt].h, bf[nt].h, acc[mt][nt]);

        __syncthreads();                // readers of buf done before it is
                                        // refilled at iteration k0+64
    }

    // ---- epilogue: C row = i + 8*half, col = l16 (per ISA C-layout) ----
#pragma unroll
    for (int mt = 0; mt < 4; ++mt) {
#pragma unroll
        for (int nt = 0; nt < 2; ++nt) {
            const int cgl = colBase + wn + nt * 16 + l16;
            const float bv = bias[cgl];
#pragma unroll
            for (int i = 0; i < 8; ++i) {
                const int r = rowBase + wm + mt * 16 + i + 8 * half;
                float v = acc[mt][nt][i] + bv;
                if constexpr (EPI == EPI_QK) {
                    // scatter to [B, H, S, HD] f16
                    int b = r >> 10, s = r & 1023;
                    int h = cgl >> 7, hd = cgl & 127;
                    ((_Float16*)outp)[(((size_t)(b * Hc + h)) * Sc + s) * HDc + hd] =
                        (_Float16)v;
                } else if constexpr (EPI == EPI_VT) {
                    // scatter transposed to [B, H, HD, S] f16
                    int b = r >> 10, s = r & 1023;
                    int h = cgl >> 7, hd = cgl & 127;
                    ((_Float16*)outp)[(((size_t)(b * Hc + h)) * HDc + hd) * Sc + s] =
                        (_Float16)v;
                } else if constexpr (EPI == EPI_RES_F32) {
                    ((float*)outp)[(size_t)r * N + cgl] =
                        v + resid[(size_t)r * N + cgl];
                } else { // EPI_GELU_F16: exact gelu
                    float g = 0.5f * v * (1.0f + erff(v * 0.70710678118654752f));
                    ((_Float16*)outp)[(size_t)r * N + cgl] = (_Float16)g;
                }
            }
        }
    }
}

// ---------------------------------------------------------------------------
// Flash attention: one wave per 16-query tile, online softmax in registers.
// q,k: [B,H,S,HD] f16.  vT: [B,H,HD,S] f16.  ctx out: [B,S,D] f16.
// ---------------------------------------------------------------------------
__global__ __launch_bounds__(128)
void attention_kernel(const _Float16* __restrict__ q16,
                      const _Float16* __restrict__ k16,
                      const _Float16* __restrict__ vT16,
                      const float* __restrict__ mask,      // [B,S]
                      _Float16* __restrict__ ctx16) {
    __shared__ _Float16 lp[4][16 * LPITCH];   // per-wave P-tile bounce

    const int wave = threadIdx.x >> 5;
    const int lane = threadIdx.x & 31;
    const int half = lane >> 4;
    const int l16  = lane & 15;

    const int tile = blockIdx.x * 4 + wave;   // 0..4095
    const int bh   = tile >> 6;               // b*H + h   (S/16 == 64)
    const int st   = tile & 63;               // query tile within sequence
    const int b    = bh >> 4;
    const int h    = bh & 15;

    const _Float16* qp = q16 + ((size_t)bh * Sc + st * 16) * HDc;
    const _Float16* kp = k16 + (size_t)bh * Sc * HDc;
    const _Float16* vp = vT16 + (size_t)bh * HDc * Sc;
    const float*    mp = mask + (size_t)b * Sc;
    _Float16* pl = lp[wave];

    // Q fragments: 4 K-chunks of the 16x128 query tile
    Frag16 qf[4];
#pragma unroll
    for (int dc = 0; dc < 4; ++dc) {
        const _Float16* src = qp + (size_t)l16 * HDc + dc * 32 + 8 * half;
        qf[dc].q[0] = *(const uint4*)src;
        qf[dc].q[1] = *(const uint4*)(src + 16);
    }

    v8f acc[8];
#pragma unroll
    for (int n = 0; n < 8; ++n) acc[n] = (v8f){0,0,0,0,0,0,0,0};
    float mrow[8], lrow[8];
#pragma unroll
    for (int i = 0; i < 8; ++i) { mrow[i] = -3.0e38f; lrow[i] = 0.0f; }

    for (int kc = 0; kc < Sc; kc += 32) {
        // ---- scores: two 16x16 tiles over this 32-key chunk ----
        v8f s0 = (v8f){0,0,0,0,0,0,0,0};
        v8f s1 = (v8f){0,0,0,0,0,0,0,0};
#pragma unroll
        for (int dc = 0; dc < 4; ++dc) {
            Frag16 b0, b1;
            const _Float16* k0p = kp + (size_t)(kc + l16) * HDc + dc * 32 + 8 * half;
            const _Float16* k1p = kp + (size_t)(kc + 16 + l16) * HDc + dc * 32 + 8 * half;
            b0.q[0] = *(const uint4*)k0p;  b0.q[1] = *(const uint4*)(k0p + 16);
            b1.q[0] = *(const uint4*)k1p;  b1.q[1] = *(const uint4*)(k1p + 16);
            s0 = wmma_f16(qf[dc].h, b0.h, s0);
            s1 = wmma_f16(qf[dc].h, b1.h, s1);
        }
        const float mk0 = mp[kc + l16];
        const float mk1 = mp[kc + 16 + l16];

        __syncthreads();               // prior P reads done before overwrite
#pragma unroll
        for (int i = 0; i < 8; ++i) {
            float a0 = s0[i] * ATT_SCALE + mk0;
            float a1 = s1[i] * ATT_SCALE + mk1;
            float mc = halfmax(fmaxf(a0, a1));
            float mn = fmaxf(mrow[i], mc);
            float corr = __expf(mrow[i] - mn);
            float p0 = __expf(a0 - mn);
            float p1 = __expf(a1 - mn);
            lrow[i] = lrow[i] * corr + halfsum(p0 + p1);
            mrow[i] = mn;
#pragma unroll
            for (int n = 0; n < 8; ++n) acc[n][i] *= corr;   // same row mapping
            const int row = i + 8 * half;
            pl[row * LPITCH + l16]      = (_Float16)p0;
            pl[row * LPITCH + 16 + l16] = (_Float16)p1;
        }
        __syncthreads();

        // P tile (16x32) as A-fragment
        Frag16 pf;
        pf.q[0] = *(const uint4*)&pl[l16 * LPITCH + 8 * half];
        pf.q[1] = *(const uint4*)&pl[l16 * LPITCH + 16 + 8 * half];

        // ctx += P * V : 8 d-subtiles of 16, V^T rows contiguous along keys
#pragma unroll
        for (int n = 0; n < 8; ++n) {
            Frag16 vb;
            const _Float16* vs = vp + (size_t)(n * 16 + l16) * Sc + kc + 8 * half;
            vb.q[0] = *(const uint4*)vs;
            vb.q[1] = *(const uint4*)(vs + 16);
            acc[n] = wmma_f16(pf.h, vb.h, acc[n]);
        }
    }

    // ---- normalize and scatter to [B, S, D] ----
#pragma unroll
    for (int n = 0; n < 8; ++n) {
#pragma unroll
        for (int i = 0; i < 8; ++i) {
            float ov = acc[n][i] / lrow[i];
            int s = st * 16 + i + 8 * half;
            int d = h * HDc + n * 16 + l16;
            ctx16[((size_t)b * Sc + s) * Dc + d] = (_Float16)ov;
        }
    }
}

// ---------------------------------------------------------------------------
// Row LayerNorm over D=2048.  One 256-thread block per row.
// DUAL: write f32 (residual path) and f16 (next GEMM A-matrix).
// ---------------------------------------------------------------------------
template <bool DUAL>
__global__ __launch_bounds__(256)
void layernorm_kernel(const float* __restrict__ in,
                      const float* __restrict__ w, const float* __restrict__ bb,
                      float* __restrict__ out32, _Float16* __restrict__ out16) {
    __shared__ float red[256];
    const int t = threadIdx.x;
    const float* x = in + (size_t)blockIdx.x * Dc;

    float s = 0.0f;
    for (int i = t; i < Dc; i += 256) s += x[i];
    red[t] = s; __syncthreads();
    for (int wdt = 128; wdt > 0; wdt >>= 1) {
        if (t < wdt) red[t] += red[t + wdt];
        __syncthreads();
    }
    const float mu = red[0] * (1.0f / Dc);
    __syncthreads();

    float s2 = 0.0f;
    for (int i = t; i < Dc; i += 256) { float d = x[i] - mu; s2 += d * d; }
    red[t] = s2; __syncthreads();
    for (int wdt = 128; wdt > 0; wdt >>= 1) {
        if (t < wdt) red[t] += red[t + wdt];
        __syncthreads();
    }
    const float inv = rsqrtf(red[0] * (1.0f / Dc) + LN_EPS);

    for (int i = t; i < Dc; i += 256) {
        float y = (x[i] - mu) * inv * w[i] + bb[i];
        out32[(size_t)blockIdx.x * Dc + i] = y;
        if constexpr (DUAL)
            out16[(size_t)blockIdx.x * Dc + i] = (_Float16)y;
    }
}

// ---------------------------------------------------------------------------
// Host launch
// ---------------------------------------------------------------------------
extern "C" void kernel_launch(void* const* d_in, const int* in_sizes, int n_in,
                              void* d_out, int out_size, void* d_ws, size_t ws_size,
                              hipStream_t stream) {
    const float* x    = (const float*)d_in[0];
    const float* mask = (const float*)d_in[1];
    const float* wq = (const float*)d_in[2];  const float* bq = (const float*)d_in[3];
    const float* wk = (const float*)d_in[4];  const float* bk = (const float*)d_in[5];
    const float* wv = (const float*)d_in[6];  const float* bv = (const float*)d_in[7];
    const float* wo = (const float*)d_in[8];  const float* bo = (const float*)d_in[9];
    const float* ln1w = (const float*)d_in[10]; const float* ln1b = (const float*)d_in[11];
    const float* wi = (const float*)d_in[12]; const float* bi = (const float*)d_in[13];
    const float* wo2 = (const float*)d_in[14]; const float* bo2 = (const float*)d_in[15];
    const float* ln2w = (const float*)d_in[16]; const float* ln2b = (const float*)d_in[17];

    char* ws = (char*)d_ws;
    size_t off = 0;
    auto carve = [&](size_t bytes) -> void* {
        void* p = ws + off;
        off += (bytes + 255) & ~(size_t)255;
        return p;
    };
    _Float16* x16   = (_Float16*)carve((size_t)Mtot * Dc * 2);
    _Float16* wq16  = (_Float16*)carve((size_t)Dc * Dc * 2);
    _Float16* wk16  = (_Float16*)carve((size_t)Dc * Dc * 2);
    _Float16* wv16  = (_Float16*)carve((size_t)Dc * Dc * 2);
    _Float16* wo16  = (_Float16*)carve((size_t)Dc * Dc * 2);
    _Float16* wi16  = (_Float16*)carve((size_t)Fc * Dc * 2);
    _Float16* wo216 = (_Float16*)carve((size_t)Dc * Fc * 2);
    _Float16* q16   = (_Float16*)carve((size_t)Mtot * Dc * 2);
    _Float16* k16   = (_Float16*)carve((size_t)Mtot * Dc * 2);
    _Float16* vT16  = (_Float16*)carve((size_t)Mtot * Dc * 2);
    _Float16* ctx16 = (_Float16*)carve((size_t)Mtot * Dc * 2);
    float*    attnp = (float*)carve((size_t)Mtot * Dc * 4);
    _Float16* h1_16 = (_Float16*)carve((size_t)Mtot * Dc * 2);
    float*    h1_32 = (float*)carve((size_t)Mtot * Dc * 4);
    _Float16* ff16  = (_Float16*)carve((size_t)Mtot * Fc * 2);
    float*    o2p   = (float*)carve((size_t)Mtot * Dc * 4);

    auto cvt = [&](const float* src, _Float16* dst, size_t n) {
        int blocks = (int)((n + 1023) / 1024);
        cvt_f32_to_f16<<<blocks, 256, 0, stream>>>(src, dst, n);
    };
    cvt(x,   x16,   (size_t)Mtot * Dc);
    cvt(wq,  wq16,  (size_t)Dc * Dc);
    cvt(wk,  wk16,  (size_t)Dc * Dc);
    cvt(wv,  wv16,  (size_t)Dc * Dc);
    cvt(wo,  wo16,  (size_t)Dc * Dc);
    cvt(wi,  wi16,  (size_t)Fc * Dc);
    cvt(wo2, wo216, (size_t)Dc * Fc);

    const dim3 blk(256);
    const dim3 gD(Dc / 128, Mtot / 128);   // (16, 32)
    const dim3 gF(Fc / 128, Mtot / 128);   // (64, 32)

    // QKV projections
    gemm_nt<EPI_QK><<<gD, blk, 0, stream>>>(x16, wq16, bq, nullptr, q16, Dc, Dc);
    gemm_nt<EPI_QK><<<gD, blk, 0, stream>>>(x16, wk16, bk, nullptr, k16, Dc, Dc);
    gemm_nt<EPI_VT><<<gD, blk, 0, stream>>>(x16, wv16, bv, nullptr, vT16, Dc, Dc);

    // Attention (4096 query tiles, 4 waves per block)
    attention_kernel<<<Mtot * Hc / 16 / 4, 128, 0, stream>>>(q16, k16, vT16, mask, ctx16);

    // Output projection + bias + residual(x) -> f32, then LN1 (dual output)
    gemm_nt<EPI_RES_F32><<<gD, blk, 0, stream>>>(ctx16, wo16, bo, x, attnp, Dc, Dc);
    layernorm_kernel<true><<<Mtot, 256, 0, stream>>>(attnp, ln1w, ln1b, h1_32, h1_16);

    // FFN: intermediate + GELU -> f16, then output + bias + residual(h1) -> f32
    gemm_nt<EPI_GELU_F16><<<gF, blk, 0, stream>>>(h1_16, wi16, bi, nullptr, ff16, Fc, Dc);
    gemm_nt<EPI_RES_F32><<<gD, blk, 0, stream>>>(ff16, wo216, bo2, h1_32, o2p, Dc, Fc);

    // LN2 -> final f32 output
    layernorm_kernel<false><<<Mtot, 256, 0, stream>>>(o2p, ln2w, ln2b, (float*)d_out, nullptr);

    (void)in_sizes; (void)n_in; (void)out_size; (void)ws_size;
}
